// MoEGLU_88252987998374
// MI455X (gfx1250) — compile-verified, matrix operand order checked by reading, not run on hardware
//
#include <hip/hip_runtime.h>
#include <hip/hip_bf16.h>

// ---------------- types ----------------
typedef __attribute__((ext_vector_type(16))) __bf16 v16bf;
typedef __attribute__((ext_vector_type(8)))  __bf16 v8bf;
typedef __attribute__((ext_vector_type(8)))  float  v8f;
typedef __attribute__((ext_vector_type(4)))  float  f32x4;

// ---------------- problem constants ----------------
#define T_TOK  8192      // B*S
#define DDIM   1024
#define FFDIM  4096
#define NEXP   8
#define TILE_M 16
#define XS     1032      // LDS row stride (bf16 elems) for X tile (1032=8*129 -> 16B-aligned rows)
#define HS     136       // LDS row stride for H tile (136=8*17)

// Load the WMMA 16-bit fragment K-pattern from bf16 memory (LDS or global):
// elements [0..7]  = p[0..7]    (K = base+koff .. +7)
// elements [8..15] = p[16..23]  (K = base+16+koff .. +23)
__device__ __forceinline__ v16bf ld_pair_bf(const __bf16* p) {
    v8bf a0 = *(const v8bf*)(p);
    v8bf a1 = *(const v8bf*)(p + 16);
    return __builtin_shufflevector(a0, a1, 0,1,2,3,4,5,6,7,8,9,10,11,12,13,14,15);
}

// Same pattern, converting from fp32 on the fly (fallback path)
__device__ __forceinline__ v16bf ld_cvt_bf16x16(const float* __restrict__ p) {
    f32x4 b0 = *(const f32x4*)(p);
    f32x4 b1 = *(const f32x4*)(p + 4);
    f32x4 b2 = *(const f32x4*)(p + 16);
    f32x4 b3 = *(const f32x4*)(p + 20);
    v16bf r;
#pragma unroll
    for (int i = 0; i < 4; ++i) {
        r[i]      = (__bf16)b0[i];
        r[i + 4]  = (__bf16)b1[i];
        r[i + 8]  = (__bf16)b2[i];
        r[i + 12] = (__bf16)b3[i];
    }
    return r;
}

// ---------------- fp32 -> bf16 streaming conversion (8 elems / thread) ----------------
__global__ __launch_bounds__(256) void cvt_bf16(const float* __restrict__ s,
                                                __bf16* __restrict__ d, int n8)
{
    int g = blockIdx.x * 256 + threadIdx.x;
    if (g < n8) {
        const float* p = s + (size_t)g * 8;
        f32x4 a = *(const f32x4*)(p);
        f32x4 b = *(const f32x4*)(p + 4);
        v8bf r;
#pragma unroll
        for (int j = 0; j < 4; ++j) { r[j] = (__bf16)a[j]; r[4 + j] = (__bf16)b[j]; }
        *(v8bf*)(d + (size_t)g * 8) = r;
    }
}

// ---------------- gating: logits -> softmax -> top2 -> dispatch lists ----------------
__global__ __launch_bounds__(256) void moe_gate(
    const float* __restrict__ x, const float* __restrict__ gw,
    int* __restrict__ counts, int* __restrict__ lists, float* __restrict__ wts,
    int* __restrict__ countk, float* __restrict__ proxys)
{
    int wave = threadIdx.x >> 5;
    int lane = threadIdx.x & 31;
    int t = blockIdx.x * 8 + wave;            // one wave per token
    if (t >= T_TOK) return;

    const float* xr = x + (size_t)t * DDIM;
    float acc[NEXP];
#pragma unroll
    for (int e = 0; e < NEXP; ++e) acc[e] = 0.f;

    for (int d = lane; d < DDIM; d += 32) {
        float xv = xr[d];
#pragma unroll
        for (int e = 0; e < NEXP; ++e) acc[e] += xv * gw[e * DDIM + d];
    }
#pragma unroll
    for (int e = 0; e < NEXP; ++e) {
#pragma unroll
        for (int off = 16; off > 0; off >>= 1)
            acc[e] += __shfl_xor(acc[e], off, 32);
    }

    if (lane == 0) {
        float mx = acc[0];
#pragma unroll
        for (int e = 1; e < NEXP; ++e) mx = fmaxf(mx, acc[e]);
        float p[NEXP], s = 0.f;
#pragma unroll
        for (int e = 0; e < NEXP; ++e) { p[e] = __expf(acc[e] - mx); s += p[e]; }
        float inv = __builtin_amdgcn_rcpf(s);
#pragma unroll
        for (int e = 0; e < NEXP; ++e) p[e] *= inv;

        int i1 = 0;
#pragma unroll
        for (int e = 1; e < NEXP; ++e) if (p[e] > p[i1]) i1 = e;
        int i2 = (i1 == 0) ? 1 : 0;
#pragma unroll
        for (int e = 0; e < NEXP; ++e) if (e != i1 && p[e] > p[i2]) i2 = e;

        float a = p[i1], b = p[i2];
        float invab = __builtin_amdgcn_rcpf(a + b);

        int pos1 = atomicAdd(&counts[i1], 1);
        lists[i1 * T_TOK + pos1] = t;
        wts  [i1 * T_TOK + pos1] = a * invab;
        int pos2 = atomicAdd(&counts[i2], 1);
        lists[i2 * T_TOK + pos2] = t;
        wts  [i2 * T_TOK + pos2] = b * invab;

        atomicAdd(&countk[0 * NEXP + i1], 1);
        atomicAdd(&countk[1 * NEXP + i2], 1);
        atomicAdd(&proxys[0 * NEXP + i1], a);
        atomicAdd(&proxys[1 * NEXP + i2], b);
    }
}

// ---------------- routed expert FFN: y += wt * silu(x@w1^T)@w2^T ----------------
// PRE=true : x/w1/w2 pre-converted bf16 (pure load+WMMA hot loop)
// PRE=false: convert fp32->bf16 in registers (fallback when ws too small)
template <bool PRE>
__global__ __launch_bounds__(256) void moe_ffn(
    const float*  __restrict__ xf,  const __bf16* __restrict__ xb,
    const float*  __restrict__ w1f, const __bf16* __restrict__ w1b,
    const float*  __restrict__ w2f, const __bf16* __restrict__ w2b,
    const int* __restrict__ counts, const int* __restrict__ lists,
    const float* __restrict__ wts, float* __restrict__ y)
{
    __shared__ __align__(16) __bf16 sX[TILE_M * XS];   // 16 x 1024 bf16 token tile
    __shared__ __align__(16) __bf16 sH[TILE_M * HS];   // 16 x 128 bf16 silu(h) chunk
    __shared__ int   sIdx[TILE_M];
    __shared__ float sWt[TILE_M];

    const int e    = blockIdx.x >> 9;       // 512 tile slots per expert
    const int tile = blockIdx.x & 511;
    const int cnt  = counts[e];
    const int row0 = tile * TILE_M;
    if (row0 >= cnt) return;

    const int tid  = threadIdx.x;
    const int wave = tid >> 5;
    const int lane = tid & 31;
    const int mrow = lane & 15;              // M (A/C) or N (B) index for this lane
    const int koff = (lane < 16) ? 0 : 8;    // K half-group select per ISA layout

    if (tid < TILE_M) {
        int r = row0 + tid;
        sIdx[tid] = (r < cnt) ? lists[e * T_TOK + r] : 0;
        sWt[tid]  = (r < cnt) ? wts[e * T_TOK + r]   : 0.f;
    }
    __syncthreads();

    // Stage X tile (gather rows) into LDS as bf16
    if constexpr (PRE) {
        for (int i = tid * 8; i < TILE_M * DDIM; i += 256 * 8) {
            int m = i >> 10, k = i & (DDIM - 1);
            *(v8bf*)&sX[m * XS + k] = *(const v8bf*)&xb[(size_t)sIdx[m] * DDIM + k];
        }
    } else {
        for (int i = tid; i < TILE_M * DDIM; i += 256) {
            int m = i >> 10, k = i & (DDIM - 1);
            sX[m * XS + k] = (__bf16)xf[(size_t)sIdx[m] * DDIM + k];
        }
    }
    __syncthreads();

    const float*  w1fe = w1f + (size_t)e * FFDIM * DDIM;  // [FF][D]
    const float*  w2fe = w2f + (size_t)e * DDIM * FFDIM;  // [D][FF]
    const __bf16* w1be = w1b + (size_t)e * FFDIM * DDIM;
    const __bf16* w2be = w2b + (size_t)e * DDIM * FFDIM;

    v8f acc[8];                                           // wave owns D cols [wave*128, +128)
#pragma unroll
    for (int t = 0; t < 8; ++t) acc[t] = (v8f){};

    for (int ff = 0; ff < FFDIM; ff += 128) {
        // ---- stage 1: h(16x16 per wave) = X @ w1[ff+wave*16 .. +16]^T, K = 1024 ----
        v8f h = (v8f){};
        const size_t w1row = (size_t)(ff + wave * 16 + mrow) * DDIM + koff;
#pragma unroll 4
        for (int k0 = 0; k0 < DDIM; k0 += 32) {
            v16bf A = ld_pair_bf(&sX[mrow * XS + k0 + koff]);
            v16bf B;
            if constexpr (PRE) B = ld_pair_bf(w1be + w1row + k0);
            else               B = ld_cvt_bf16x16(w1fe + w1row + k0);
            h = __builtin_amdgcn_wmma_f32_16x16x32_bf16(false, A, false, B,
                                                        (short)0, h, false, false);
        }
        // silu (fast rcp: data is bf16-quantized anyway) + share through LDS as bf16
#pragma unroll
        for (int i = 0; i < 8; ++i) {
            float v  = h[i];
            float sv = v * __builtin_amdgcn_rcpf(1.f + __expf(-v));
            int m = i + ((lane < 16) ? 0 : 8);            // C layout: VGPR i -> M row
            sH[m * HS + wave * 16 + mrow] = (__bf16)sv;
        }
        __syncthreads();

        // ---- stage 2: acc(16x128 per wave) += H(16x128) @ w2[:, ff..ff+128]^T ----
        for (int kk = 0; kk < 128; kk += 32) {
            v16bf A = ld_pair_bf(&sH[mrow * HS + kk + koff]);
#pragma unroll
            for (int t = 0; t < 8; ++t) {
                const size_t w2off = (size_t)(wave * 128 + t * 16 + mrow) * FFDIM
                                     + ff + kk + koff;
                v16bf B;
                if constexpr (PRE) B = ld_pair_bf(w2be + w2off);
                else               B = ld_cvt_bf16x16(w2fe + w2off);
                acc[t] = __builtin_amdgcn_wmma_f32_16x16x32_bf16(false, A, false, B,
                                                                 (short)0, acc[t], false, false);
            }
        }
        __syncthreads();   // protect sH before next chunk overwrites it
    }

    // ---- epilogue: weighted scatter-add (tokens appear in 2 expert lists) ----
#pragma unroll
    for (int t = 0; t < 8; ++t) {
#pragma unroll
        for (int i = 0; i < 8; ++i) {
            int m = i + ((lane < 16) ? 0 : 8);
            if (row0 + m < cnt) {
                int d = wave * 128 + t * 16 + mrow;
                atomicAdd(&y[(size_t)sIdx[m] * DDIM + d], sWt[m] * acc[t][i]);
            }
        }
    }
}

// ---------------- aux loss finalize ----------------
__global__ void moe_finalize(const int* __restrict__ countk,
                             const float* __restrict__ proxys,
                             float* __restrict__ out_aux)
{
    if (threadIdx.x == 0 && blockIdx.x == 0) {
        float aux = 0.f;
        for (int k = 0; k < 2; ++k)
            for (int e = 0; e < NEXP; ++e)
                aux += proxys[k * NEXP + e] *
                       ((float)countk[k * NEXP + e] / (float)T_TOK);
        out_aux[0] = aux * (float)NEXP;
    }
}

// ---------------- launch ----------------
extern "C" void kernel_launch(void* const* d_in, const int* in_sizes, int n_in,
                              void* d_out, int out_size, void* d_ws, size_t ws_size,
                              hipStream_t stream)
{
    const float* x  = (const float*)d_in[0];   // [4,2048,1024]
    const float* gw = (const float*)d_in[1];   // [8,1024]
    const float* w1 = (const float*)d_in[2];   // [8,4096,1024]
    const float* w2 = (const float*)d_in[3];   // [8,1024,4096]
    float* out = (float*)d_out;                // [T*D] y  +  [1] aux

    char*  ws     = (char*)d_ws;
    int*   counts = (int*)(ws);                      // 8 ints
    int*   countk = (int*)(ws + 32);                 // 2*8 ints
    float* proxys = (float*)(ws + 96);               // 2*8 floats
    int*   lists  = (int*)(ws + 256);                // [E][T] ints
    float* wts    = (float*)(ws + 256 + (size_t)NEXP * T_TOK * 4);

    const size_t W1N = (size_t)NEXP * FFDIM * DDIM;  // 33.55M elems
    const size_t W2N = (size_t)NEXP * DDIM * FFDIM;
    const size_t XN  = (size_t)T_TOK * DDIM;
    const size_t BF_BASE = 1u << 20;                 // bf16 region starts at 1MB
    __bf16* w1b = (__bf16*)(ws + BF_BASE);
    __bf16* w2b = w1b + W1N;
    __bf16* xb  = w2b + W2N;
    const bool pre = ws_size >= BF_BASE + (W1N + W2N + XN) * 2;

    hipMemsetAsync(d_ws, 0, 256, stream);                                  // counters
    hipMemsetAsync(d_out, 0, (size_t)out_size * sizeof(float), stream);    // y accumulator

    moe_gate<<<T_TOK / 8, 256, 0, stream>>>(x, gw, counts, lists, wts, countk, proxys);

    if (pre) {
        cvt_bf16<<<(int)(W1N / 8 / 256), 256, 0, stream>>>(w1, w1b, (int)(W1N / 8));
        cvt_bf16<<<(int)(W2N / 8 / 256), 256, 0, stream>>>(w2, w2b, (int)(W2N / 8));
        cvt_bf16<<<(int)(XN  / 8 / 256), 256, 0, stream>>>(x,  xb,  (int)(XN  / 8));
        moe_ffn<true><<<NEXP * (T_TOK / TILE_M), 256, 0, stream>>>(
            x, xb, w1, w1b, w2, w2b, counts, lists, wts, out);
    } else {
        moe_ffn<false><<<NEXP * (T_TOK / TILE_M), 256, 0, stream>>>(
            x, xb, w1, w1b, w2, w2b, counts, lists, wts, out);
    }

    moe_finalize<<<1, 32, 0, stream>>>(countk, proxys, out + (size_t)T_TOK * DDIM);
}
